// SparseMoE_9921374454345
// MI455X (gfx1250) — matrix operand order repaired and checked by reading, not compiled, not run on hardware
//
#include <hip/hip_runtime.h>
#include <hip/hip_bf16.h>
#include <stdint.h>
#include <math.h>

// ---------------------------------------------------------------------------
// Sparse MoE (top-2 of 8 experts), SwiGLU, for MI455X (gfx1250, wave32, WMMA)
//
// T=4096 tokens, D=1024, E=8, H=2816, top-2.  Sparse FLOPs ~142 G;
// mandatory HBM traffic ~380 MB fp32 (weights > L2) => ~16 us floor @23.3TB/s.
// All three GEMMs run on v_wmma_f32_16x16x32_bf16; fp32 operands are converted
// to bf16 while staging into LDS.  GEMM2's A tiles (bf16 SwiGLU intermediate)
// are moved with global_load_async_to_lds_b128 (ASYNCcnt).
//
// Workspace (ws_size must be >= ~47 MB):
//   cw [T*E] f32, tope [T*2] i32, counts/cursors i32, offsets i32,
//   list [2T] i32, Hbuf [2T, H] bf16.
// ---------------------------------------------------------------------------

#define T_TOK  4096
#define DMODEL 1024
#define NEXP   8
#define HDIM   2816
#define TOPK   2

#define BM 64
#define BN 128
#define BK 64          // two WMMA K-steps per LDS stage

typedef __attribute__((ext_vector_type(16))) __bf16 v16bf;
typedef __attribute__((ext_vector_type(8)))  float  v8f;

__device__ __forceinline__ __bf16 f2bf(float f) { return (__bf16)f; }

// ---------------- zero out + counters -------------------------------------
__global__ void moe_zero(float* __restrict__ out, int* __restrict__ cnt16) {
    size_t n = (size_t)T_TOK * DMODEL;
    for (size_t i = (size_t)blockIdx.x * blockDim.x + threadIdx.x; i < n;
         i += (size_t)gridDim.x * blockDim.x)
        out[i] = 0.0f;
    if (blockIdx.x == 0 && threadIdx.x < 16) cnt16[threadIdx.x] = 0;
}

// ---------------- gating: logits, softmax, top-2, counts -------------------
__global__ void moe_gate(const float* __restrict__ x, const float* __restrict__ wg,
                         float* __restrict__ cw, int* __restrict__ tope,
                         int* __restrict__ counts) {
    int wave = threadIdx.x >> 5;
    int lane = threadIdx.x & 31;
    int t = blockIdx.x * 8 + wave;
    if (t >= T_TOK) return;

    float acc[NEXP];
#pragma unroll
    for (int e = 0; e < NEXP; ++e) acc[e] = 0.0f;

    const float* xrow = x + (size_t)t * DMODEL;
    for (int d = lane; d < DMODEL; d += 32) {
        float xv = xrow[d];
        const float* wrow = wg + (size_t)d * NEXP;
#pragma unroll
        for (int e = 0; e < NEXP; ++e) acc[e] += xv * wrow[e];
    }
#pragma unroll
    for (int off = 16; off > 0; off >>= 1) {
#pragma unroll
        for (int e = 0; e < NEXP; ++e) acc[e] += __shfl_xor(acc[e], off, 32);
    }
    if (lane == 0) {
        int e1 = 0; float m1 = acc[0];
        for (int e = 1; e < NEXP; ++e) if (acc[e] > m1) { m1 = acc[e]; e1 = e; }
        int e2 = (e1 == 0) ? 1 : 0; float m2 = acc[e2];
        for (int e = 0; e < NEXP; ++e)
            if (e != e1 && acc[e] > m2) { m2 = acc[e]; e2 = e; }
        float p2 = __expf(m2 - m1);
        float c1 = 1.0f / (1.0f + p2);
        float c2 = p2 * c1;
        float* cwrow = cw + (size_t)t * NEXP;
#pragma unroll
        for (int e = 0; e < NEXP; ++e) cwrow[e] = 0.0f;
        cwrow[e1] = c1; cwrow[e2] = c2;
        tope[t * 2 + 0] = e1; tope[t * 2 + 1] = e2;
        atomicAdd(&counts[e1], 1);
        atomicAdd(&counts[e2], 1);
    }
}

// ---------------- 8-entry exclusive scan -----------------------------------
__global__ void moe_scan(const int* __restrict__ counts, int* __restrict__ offsets) {
    if (blockIdx.x == 0 && threadIdx.x == 0) {
        int run = 0;
        for (int e = 0; e < NEXP; ++e) { offsets[e] = run; run += counts[e]; }
        offsets[NEXP] = run;
    }
}

// ---------------- build compacted per-expert token lists -------------------
__global__ void moe_build(const int* __restrict__ tope, const int* __restrict__ offsets,
                          int* __restrict__ cursors, int* __restrict__ list) {
    int t = blockIdx.x * blockDim.x + threadIdx.x;
    if (t >= T_TOK) return;
#pragma unroll
    for (int k = 0; k < TOPK; ++k) {
        int e = tope[t * 2 + k];
        int slot = atomicAdd(&cursors[e], 1);
        list[offsets[e] + slot] = t;
    }
}

// ---------------- GEMM1: Hbuf = silu(Xg*w1) * (Xg*w3), bf16 ----------------
__global__ __launch_bounds__(256)
void moe_gemm1(const float* __restrict__ x,
               const float* __restrict__ w1, const float* __restrict__ w3,
               const int* __restrict__ offsets, const int* __restrict__ list,
               __bf16* __restrict__ hbuf) {
    int e = blockIdx.z;
    int off = offsets[e];
    int count = offsets[e + 1] - off;
    int r0 = blockIdx.y * BM;
    if (r0 >= count) return;
    int n0 = blockIdx.x * BN;

    __shared__ __bf16 As[BM][BK + 8];      // row-major [m][k]
    __shared__ __bf16 B1s[BN][BK + 8];     // fragment-major [n][k]
    __shared__ __bf16 B3s[BN][BK + 8];
    __shared__ int tok[BM];

    int tid = threadIdx.x;
    if (tid < BM) {
        int r = r0 + tid;
        tok[tid] = (r < count) ? list[off + r] : 0;
    }
    __syncthreads();

    int lane  = tid & 31;
    int wv    = tid >> 5;
    int wm    = wv >> 2;          // 0..1  (row half)
    int wn    = wv & 3;           // 0..3  (col quarter)
    int lrow  = lane & 15;
    int khalf = lane >> 4;

    v8f zero = {};
    v8f acc1[2][2], acc3[2][2];
#pragma unroll
    for (int i = 0; i < 2; ++i)
#pragma unroll
        for (int j = 0; j < 2; ++j) { acc1[i][j] = zero; acc3[i][j] = zero; }

    // per-thread staging coordinates
    int ar = tid >> 2;              // 0..63 (token row)
    int ac = (tid & 3) * 16;        // 0,16,32,48
    int bk = tid >> 2;              // 0..63 (k row)
    int bn = (tid & 3) * 32;        // 0,32,64,96

    for (int k0 = 0; k0 < DMODEL; k0 += BK) {
        __syncthreads();
        {   // A: 64x64 gathered x rows, fp32 -> bf16 (16 elems/thread)
            const float* src = x + (size_t)tok[ar] * DMODEL + k0 + ac;
#pragma unroll
            for (int i = 0; i < 16; ++i) As[ar][ac + i] = f2bf(src[i]);
            if (k0 + BK < DMODEL) __builtin_prefetch(src + BK, 0, 3);
        }
        {   // B1/B3: 64x128 each, transposed into [n][k] (32 elems/thread each)
            const float* s1 = w1 + ((size_t)e * DMODEL + (k0 + bk)) * HDIM + n0 + bn;
            const float* s3 = w3 + ((size_t)e * DMODEL + (k0 + bk)) * HDIM + n0 + bn;
#pragma unroll
            for (int i = 0; i < 32; ++i) {
                B1s[bn + i][bk] = f2bf(s1[i]);
                B3s[bn + i][bk] = f2bf(s3[i]);
            }
            if (k0 + BK < DMODEL) {
                __builtin_prefetch(s1 + (size_t)BK * HDIM, 0, 3);
                __builtin_prefetch(s3 + (size_t)BK * HDIM, 0, 3);
            }
        }
        __syncthreads();

#pragma unroll
        for (int ks = 0; ks < BK; ks += 32) {
            v16bf afr[2];
#pragma unroll
            for (int tm = 0; tm < 2; ++tm) {
                int row = wm * 32 + tm * 16 + lrow;
#pragma unroll
                for (int i = 0; i < 8; ++i) afr[tm][i]     = As[row][ks + khalf * 8 + i];
#pragma unroll
                for (int i = 0; i < 8; ++i) afr[tm][8 + i] = As[row][ks + 16 + khalf * 8 + i];
            }
#pragma unroll
            for (int tn = 0; tn < 2; ++tn) {
                int col = wn * 32 + tn * 16 + lrow;
                v16bf b1, b3;
#pragma unroll
                for (int i = 0; i < 16; ++i) b1[i] = B1s[col][ks + (i >> 3) * 16 + khalf * 8 + (i & 7)];
#pragma unroll
                for (int i = 0; i < 16; ++i) b3[i] = B3s[col][ks + (i >> 3) * 16 + khalf * 8 + (i & 7)];
#pragma unroll
                for (int tm = 0; tm < 2; ++tm) {
                    acc1[tm][tn] = __builtin_amdgcn_wmma_f32_16x16x32_bf16(
                        false, afr[tm], false, b1, (short)0, acc1[tm][tn], false, false);
                    acc3[tm][tn] = __builtin_amdgcn_wmma_f32_16x16x32_bf16(
                        false, afr[tm], false, b3, (short)0, acc3[tm][tn], false, false);
                }
            }
        }
    }

    // epilogue: SwiGLU (fast exp + rcp), store bf16 to Hbuf
#pragma unroll
    for (int tm = 0; tm < 2; ++tm) {
#pragma unroll
        for (int tn = 0; tn < 2; ++tn) {
#pragma unroll
            for (int r = 0; r < 8; ++r) {
                int row = wm * 32 + tm * 16 + khalf * 8 + r;
                int col = wn * 32 + tn * 16 + lrow;
                if (r0 + row < count) {
                    float g = acc1[tm][tn][r];
                    float u = acc3[tm][tn][r];
                    float h = g * __builtin_amdgcn_rcpf(1.0f + __expf(-g)) * u;
                    hbuf[(size_t)(off + r0 + row) * HDIM + n0 + col] = f2bf(h);
                }
            }
        }
    }
}

// ---------------- GEMM2: out += cw * (Hbuf * w2), atomic scatter -----------
__global__ __launch_bounds__(256)
void moe_gemm2(const __bf16* __restrict__ hbuf, const float* __restrict__ w2,
               const int* __restrict__ offsets, const int* __restrict__ list,
               const float* __restrict__ cw, float* __restrict__ out) {
    int e = blockIdx.z;
    int off = offsets[e];
    int count = offsets[e + 1] - off;
    int r0 = blockIdx.y * BM;
    if (r0 >= count) return;
    int n0 = blockIdx.x * BN;

    __shared__ __bf16 As[BM][BK + 8];
    __shared__ __bf16 Bs[BN][BK + 8];
    __shared__ int   tok[BM];
    __shared__ float cws[BM];

    int tid = threadIdx.x;
    if (tid < BM) {
        int r = r0 + tid;
        int t = (r < count) ? list[off + r] : -1;
        tok[tid] = t;
        cws[tid] = (t >= 0) ? cw[(size_t)t * NEXP + e] : 0.0f;
    }
    __syncthreads();

    int lane  = tid & 31;
    int wv    = tid >> 5;
    int wm    = wv >> 2;
    int wn    = wv & 3;
    int lrow  = lane & 15;
    int khalf = lane >> 4;

    v8f zero = {};
    v8f acc[2][2];
#pragma unroll
    for (int i = 0; i < 2; ++i)
#pragma unroll
        for (int j = 0; j < 2; ++j) acc[i][j] = zero;

    // LDS byte base of As (flat LDS pointers carry the offset in the low dword)
    uint32_t ldsA = (uint32_t)(uintptr_t)(&As[0][0]);

    int bk = tid >> 2;              // 0..63
    int bn = (tid & 3) * 32;        // 0,32,64,96

    for (int k0 = 0; k0 < HDIM; k0 += BK) {
        __syncthreads();
        {   // A: 64x64 bf16 tile from Hbuf via async global->LDS (2 x 16B / thread)
#pragma unroll
            for (int i = 0; i < 2; ++i) {
                int chunk = tid + 256 * i;           // 0..511
                int r  = chunk >> 3;                 // 0..63
                int ce = (chunk & 7) * 8;            // k-element offset, 16B chunks
                int rr = r0 + r;
                int rowi = (rr < count) ? rr : (count - 1);  // clamp: tail rows guarded at epilogue
                const __bf16* g = hbuf + (size_t)(off + rowi) * HDIM + k0 + ce;
                uint32_t l = ldsA + (uint32_t)(r * (BK + 8) + ce) * 2u;
                asm volatile("global_load_async_to_lds_b128 %0, %1, off"
                             :: "v"(l), "v"((unsigned long long)(uintptr_t)g)
                             : "memory");
            }
        }
        {   // B: w2 64x128 tile, fp32 -> bf16, transposed into [n][k]
            const float* s2 = w2 + ((size_t)e * HDIM + (k0 + bk)) * DMODEL + n0 + bn;
#pragma unroll
            for (int i = 0; i < 32; ++i) Bs[bn + i][bk] = f2bf(s2[i]);
            if (k0 + BK < HDIM) __builtin_prefetch(s2 + (size_t)BK * DMODEL, 0, 3);
        }
        asm volatile("s_wait_asynccnt 0x0" ::: "memory");
        __syncthreads();

#pragma unroll
        for (int ks = 0; ks < BK; ks += 32) {
            v16bf afr[2];
#pragma unroll
            for (int tm = 0; tm < 2; ++tm) {
                int row = wm * 32 + tm * 16 + lrow;
#pragma unroll
                for (int i = 0; i < 8; ++i) afr[tm][i]     = As[row][ks + khalf * 8 + i];
#pragma unroll
                for (int i = 0; i < 8; ++i) afr[tm][8 + i] = As[row][ks + 16 + khalf * 8 + i];
            }
#pragma unroll
            for (int tn = 0; tn < 2; ++tn) {
                int col = wn * 32 + tn * 16 + lrow;
                v16bf b;
#pragma unroll
                for (int i = 0; i < 16; ++i) b[i] = Bs[col][ks + (i >> 3) * 16 + khalf * 8 + (i & 7)];
#pragma unroll
                for (int tm = 0; tm < 2; ++tm) {
                    acc[tm][tn] = __builtin_amdgcn_wmma_f32_16x16x32_bf16(
                        false, afr[tm], false, b, (short)0, acc[tm][tn], false, false);
                }
            }
        }
    }

#pragma unroll
    for (int tm = 0; tm < 2; ++tm) {
#pragma unroll
        for (int tn = 0; tn < 2; ++tn) {
#pragma unroll
            for (int r = 0; r < 8; ++r) {
                int row = wm * 32 + tm * 16 + khalf * 8 + r;
                int col = wn * 32 + tn * 16 + lrow;
                if (r0 + row < count) {
                    int t = tok[row];
                    float v = acc[tm][tn][r] * cws[row];
                    atomicAdd(&out[(size_t)t * DMODEL + n0 + col], v);
                }
            }
        }
    }
}

// ---------------------------------------------------------------------------
extern "C" void kernel_launch(void* const* d_in, const int* in_sizes, int n_in,
                              void* d_out, int out_size, void* d_ws, size_t ws_size,
                              hipStream_t stream) {
    (void)in_sizes; (void)n_in; (void)out_size; (void)ws_size;
    const float* x  = (const float*)d_in[0];
    const float* wg = (const float*)d_in[1];
    const float* w1 = (const float*)d_in[2];
    const float* w2 = (const float*)d_in[3];
    const float* w3 = (const float*)d_in[4];
    float* out = (float*)d_out;

    uint8_t* ws = (uint8_t*)d_ws;
    float* cwp   = (float*)ws;            ws += (size_t)T_TOK * NEXP * 4;
    int*   tope  = (int*)ws;              ws += (size_t)T_TOK * 2 * 4;
    int*   cnt   = (int*)ws;              ws += 64;   // counts[8] + cursors[8]
    int*   offs  = (int*)ws;              ws += 64;   // offsets[9]
    int*   list  = (int*)ws;              ws += (size_t)T_TOK * 2 * 4;
    ws = (uint8_t*)(((uintptr_t)ws + 255) & ~(uintptr_t)255);
    __bf16* hbuf = (__bf16*)ws;           // [2T, H] bf16 ~ 44 MB

    int* counts  = cnt;
    int* cursors = cnt + 8;

    moe_zero <<<512, 256, 0, stream>>>(out, cnt);
    moe_gate <<<T_TOK / 8, 256, 0, stream>>>(x, wg, cwp, tope, counts);
    moe_scan <<<1, 32, 0, stream>>>(counts, offs);
    moe_build<<<T_TOK / 256, 256, 0, stream>>>(tope, offs, cursors, list);

    dim3 g1(HDIM / BN, T_TOK / BM, NEXP);   // (22, 64, 8); excess row-tiles exit early
    moe_gemm1<<<g1, 256, 0, stream>>>(x, w1, w3, offs, list, hbuf);

    dim3 g2(DMODEL / BN, T_TOK / BM, NEXP); // (8, 64, 8)
    moe_gemm2<<<g2, 256, 0, stream>>>(hbuf, w2, offs, list, cwp, out);
}
